// MultiHeadAttentionBlock_23630910062888
// MI455X (gfx1250) — matrix operand order repaired and checked by reading, not compiled, not run on hardware
//
#include <hip/hip_runtime.h>
#include <hip/hip_bf16.h>

#define B_  2
#define S_  2048
#define D_  1024
#define H_  16
#define DK_ 64
#define EPS_ 1e-6f

typedef __attribute__((ext_vector_type(16))) _Float16 v16h;
typedef __attribute__((ext_vector_type(8)))  float    v8f;
typedef __attribute__((ext_vector_type(4)))  _Float16 h4;

// async-copy builtin operand types: pointer to 16B int vector, in the
// global (AS1) and LDS (AS3) address spaces (per hipcc diagnostic).
typedef int v4i __attribute__((vector_size(16)));
typedef __attribute__((address_space(1))) v4i* gp4i;
typedef __attribute__((address_space(3))) v4i* sp4i;

union Frag { v16h v; uint4 q[2]; };

static __device__ __forceinline__ v8f wmma_f16(v16h a, v16h b, v8f c) {
  // (neg_a, A, neg_b, B, c_mod, C, reuse_a, reuse_b)
  return __builtin_amdgcn_wmma_f32_16x16x32_f16(false, a, false, b, (short)0, c,
                                                false, false);
}

#if __has_builtin(__builtin_amdgcn_global_load_async_to_lds_b128)
#define HAVE_ASYNC_LDS 1
#else
#define HAVE_ASYNC_LDS 0
#endif

#if __has_builtin(__builtin_amdgcn_s_wait_asynccnt)
#define WAIT_ASYNC(n) __builtin_amdgcn_s_wait_asynccnt(n)
#else
#define WAIT_ASYNC(n) asm volatile("s_wait_asynccnt " #n ::: "memory")
#endif

// ---------------------------------------------------------------------------
// fp32 -> f16 streaming convert (vectorized float4 -> h4)
// ---------------------------------------------------------------------------
__global__ __launch_bounds__(256)
void cvt_f32_to_f16(const float* __restrict__ in, _Float16* __restrict__ out,
                    int n4) {
  int i = blockIdx.x * 256 + threadIdx.x;
  if (i < n4) {
    float4 f = ((const float4*)in)[i];
    h4 h = {(_Float16)f.x, (_Float16)f.y, (_Float16)f.z, (_Float16)f.w};
    ((h4*)out)[i] = h;
  }
}

// ---------------------------------------------------------------------------
// GEMM: C[M,N] = A[M,K] * W[N,K]^T + bias[N]   (f16 operands, f32 out)
// Block = 256 threads (8 waves), tile 128x128, wave tile 32x64.
// Double-buffered LDS staging via GLOBAL_LOAD_ASYNC_TO_LDS_B128 (ASYNCcnt).
// ---------------------------------------------------------------------------
__global__ __launch_bounds__(256)
void gemm_f16_wmma(const _Float16* __restrict__ A, const _Float16* __restrict__ W,
                   const float* __restrict__ bias, float* __restrict__ C,
                   int M, int N, int K) {
  __shared__ __align__(16) _Float16 sA[2][128 * 32];
  __shared__ __align__(16) _Float16 sB[2][128 * 32];
  const int tid  = threadIdx.x;
  const int lane = tid & 31;
  const int wave = tid >> 5;
  const int bm = blockIdx.x * 128;
  const int bn = blockIdx.y * 128;
  const int mw = (wave & 3) * 32;
  const int nw = (wave >> 2) * 64;
  const int r16  = lane & 15;
  const int aklo = (lane & 16) ? 8 : 0;    // A-fragment K interleave
  const int bkhi = (lane & 16) ? 16 : 0;   // unused: B fragment uses aklo as well

  // each thread stages two 16B chunks per matrix per tile (128x32 f16 = 8KB)
  const int ch0r = tid >> 2,          ch0c = tid & 3;
  const int ch1r = (tid + 256) >> 2,  ch1c = (tid + 256) & 3;

  auto issue = [&](int buf, int k0) {
#if HAVE_ASYNC_LDS
    __builtin_amdgcn_global_load_async_to_lds_b128(
        (gp4i)&A[(size_t)(bm + ch0r) * K + k0 + ch0c * 8],
        (sp4i)&sA[buf][ch0r * 32 + ch0c * 8], 0, 0);
    __builtin_amdgcn_global_load_async_to_lds_b128(
        (gp4i)&A[(size_t)(bm + ch1r) * K + k0 + ch1c * 8],
        (sp4i)&sA[buf][ch1r * 32 + ch1c * 8], 0, 0);
    __builtin_amdgcn_global_load_async_to_lds_b128(
        (gp4i)&W[(size_t)(bn + ch0r) * K + k0 + ch0c * 8],
        (sp4i)&sB[buf][ch0r * 32 + ch0c * 8], 0, 0);
    __builtin_amdgcn_global_load_async_to_lds_b128(
        (gp4i)&W[(size_t)(bn + ch1r) * K + k0 + ch1c * 8],
        (sp4i)&sB[buf][ch1r * 32 + ch1c * 8], 0, 0);
#else
    uint4 a0 = *(const uint4*)&A[(size_t)(bm + ch0r) * K + k0 + ch0c * 8];
    uint4 a1 = *(const uint4*)&A[(size_t)(bm + ch1r) * K + k0 + ch1c * 8];
    uint4 b0 = *(const uint4*)&W[(size_t)(bn + ch0r) * K + k0 + ch0c * 8];
    uint4 b1 = *(const uint4*)&W[(size_t)(bn + ch1r) * K + k0 + ch1c * 8];
    *(uint4*)&sA[buf][ch0r * 32 + ch0c * 8] = a0;
    *(uint4*)&sA[buf][ch1r * 32 + ch1c * 8] = a1;
    *(uint4*)&sB[buf][ch0r * 32 + ch0c * 8] = b0;
    *(uint4*)&sB[buf][ch1r * 32 + ch1c * 8] = b1;
#endif
  };

  const v8f z8 = {0.f, 0.f, 0.f, 0.f, 0.f, 0.f, 0.f, 0.f};
  v8f acc[2][4];
  for (int i = 0; i < 2; ++i)
    for (int j = 0; j < 4; ++j) acc[i][j] = z8;

  const int T = K / 32;
  issue(0, 0);
  for (int t = 0; t < T; ++t) {
    const int buf = t & 1;
    if (t + 1 < T) {
      issue(buf ^ 1, (t + 1) * 32);
      WAIT_ASYNC(4);              // this tile's 4 loads done; next 4 in flight
    } else {
      WAIT_ASYNC(0);
    }
    __syncthreads();              // everyone's current tile is in LDS

    Frag a[2], b[4];
#pragma unroll
    for (int mt = 0; mt < 2; ++mt) {
      int r = mw + mt * 16 + r16;
      a[mt].q[0] = *(const uint4*)&sA[buf][r * 32 + aklo];
      a[mt].q[1] = *(const uint4*)&sA[buf][r * 32 + 16 + aklo];
    }
#pragma unroll
    for (int nt = 0; nt < 4; ++nt) {
      int n = nw + nt * 16 + r16;
      b[nt].q[0] = *(const uint4*)&sB[buf][n * 32 + aklo];
      b[nt].q[1] = *(const uint4*)&sB[buf][n * 32 + 16 + aklo];
    }
#pragma unroll
    for (int mt = 0; mt < 2; ++mt)
#pragma unroll
      for (int nt = 0; nt < 4; ++nt)
        acc[mt][nt] = wmma_f16(a[mt].v, b[nt].v, acc[mt][nt]);

    __syncthreads();              // done reading buf before t+2 overwrites it
  }

#pragma unroll
  for (int nt = 0; nt < 4; ++nt) {
    int col = bn + nw + nt * 16 + r16;
    float bv = bias[col];
#pragma unroll
    for (int mt = 0; mt < 2; ++mt)
#pragma unroll
      for (int r = 0; r < 8; ++r) {
        int row = bm + mw + mt * 16 + r + ((lane >> 4) << 3);
        C[(size_t)row * N + col] = acc[mt][nt][r] + bv;
      }
  }
}

// ---------------------------------------------------------------------------
// RMSNorm + RoPE for Q,K (fp32 -> f16 [B,H,S,DK]); V pack transposed f16
// [B,H,DK,S]. One wave per (b,s,h); lane handles d and d+32.
// ---------------------------------------------------------------------------
__global__ __launch_bounds__(32)
void norm_rope_pack(const float* __restrict__ Qf, const float* __restrict__ Kf,
                    const float* __restrict__ Vf,
                    const float* __restrict__ cosT, const float* __restrict__ sinT,
                    const float* __restrict__ qw, const float* __restrict__ kw,
                    _Float16* __restrict__ Qh, _Float16* __restrict__ Kh,
                    _Float16* __restrict__ Vt) {
  const int idx = blockIdx.x;           // (b*S+s)*H + h
  const int h  = idx % H_;
  const int bs = idx / H_;
  const int s  = bs % S_;
  const int b  = bs / S_;
  const int lane = threadIdx.x;
  const size_t src = (size_t)bs * D_ + h * DK_;
  const size_t dst = ((size_t)(b * H_ + h) * S_ + s) * DK_;

  const float c1 = cosT[s * DK_ + lane],      c2 = cosT[s * DK_ + lane + 32];
  const float s1 = sinT[s * DK_ + lane],      s2 = sinT[s * DK_ + lane + 32];

  {  // Q: rmsnorm + rope
    float x1 = Qf[src + lane], x2 = Qf[src + lane + 32];
    float ss = x1 * x1 + x2 * x2;
    for (int o = 16; o >= 1; o >>= 1) ss += __shfl_xor(ss, o, 32);
    float nrm = rsqrtf(ss * (1.0f / DK_) + EPS_);
    x1 *= nrm * qw[lane];
    x2 *= nrm * qw[lane + 32];
    Qh[dst + lane]      = (_Float16)(x1 * c1 - x2 * s1);
    Qh[dst + lane + 32] = (_Float16)(x2 * c2 + x1 * s2);
  }
  {  // K: rmsnorm + rope
    float x1 = Kf[src + lane], x2 = Kf[src + lane + 32];
    float ss = x1 * x1 + x2 * x2;
    for (int o = 16; o >= 1; o >>= 1) ss += __shfl_xor(ss, o, 32);
    float nrm = rsqrtf(ss * (1.0f / DK_) + EPS_);
    x1 *= nrm * kw[lane];
    x2 *= nrm * kw[lane + 32];
    Kh[dst + lane]      = (_Float16)(x1 * c1 - x2 * s1);
    Kh[dst + lane + 32] = (_Float16)(x2 * c2 + x1 * s2);
  }
  {  // V: transpose pack [B,H,DK,S]
    const size_t vb = (size_t)(b * H_ + h) * DK_;
    Vt[(vb + lane) * S_ + s]      = (_Float16)Vf[src + lane];
    Vt[(vb + lane + 32) * S_ + s] = (_Float16)Vf[src + lane + 32];
  }
}

// ---------------------------------------------------------------------------
// Causal flash attention. One wave per 16-row query tile; 32-key steps.
// Out f16 [B,S,H*DK] (feeds the O-projection GEMM's A operand).
// ---------------------------------------------------------------------------
__global__ __launch_bounds__(256)
void flash_attn(const _Float16* __restrict__ Qh, const _Float16* __restrict__ Kh,
                const _Float16* __restrict__ Vt, _Float16* __restrict__ Out) {
  __shared__ __align__(16) _Float16 sP[8][16 * 32];  // per-wave P tile
  const int lane = threadIdx.x & 31;
  const int wave = threadIdx.x >> 5;
  const int tile = blockIdx.x * 8 + wave;  // B*H*(S/16) tiles total
  const int qt = tile & 127;               // S/16 = 128
  const int bh = tile >> 7;
  const int h  = bh % H_;
  const int b  = bh / H_;
  const int qbase = qt * 16;
  const int r16  = lane & 15;
  const int aklo = (lane & 16) ? 8 : 0;
  const int bkhi = (lane & 16) ? 16 : 0;
  const int rhalf = (lane >> 4) << 3;      // +0 / +8 row offset in C layout

  const _Float16* Qb = Qh + (size_t)bh * S_ * DK_;
  const _Float16* Kb = Kh + (size_t)bh * S_ * DK_;
  const _Float16* Vb = Vt + (size_t)bh * DK_ * S_;

  Frag qa[2];
#pragma unroll
  for (int c = 0; c < 2; ++c) {
    size_t base = (size_t)(qbase + r16) * DK_ + c * 32;
    qa[c].q[0] = *(const uint4*)&Qb[base + aklo];
    qa[c].q[1] = *(const uint4*)&Qb[base + 16 + aklo];
  }

  const v8f z8 = {0.f, 0.f, 0.f, 0.f, 0.f, 0.f, 0.f, 0.f};
  v8f acc[4];
  float mrow[8], lrow[8];
#pragma unroll
  for (int nt = 0; nt < 4; ++nt) acc[nt] = z8;
#pragma unroll
  for (int r = 0; r < 8; ++r) { mrow[r] = -1e30f; lrow[r] = 0.f; }

  const int nkt = (qbase + 15) / 32 + 1;
  const float scale = 0.125f;              // 1/sqrt(DK)

  for (int kt = 0; kt < nkt; ++kt) {
    const int kb = kt * 32;
    Frag bk[2][2];
#pragma unroll
    for (int j = 0; j < 2; ++j) {
      size_t keyrow = (size_t)(kb + j * 16 + r16) * DK_;
#pragma unroll
      for (int c = 0; c < 2; ++c) {
        bk[j][c].q[0] = *(const uint4*)&Kb[keyrow + c * 32 + aklo];
        bk[j][c].q[1] = *(const uint4*)&Kb[keyrow + c * 32 + 16 + aklo];
      }
    }
    v8f s0 = z8, s1 = z8;
    s0 = wmma_f16(qa[0].v, bk[0][0].v, s0);
    s0 = wmma_f16(qa[1].v, bk[0][1].v, s0);
    s1 = wmma_f16(qa[0].v, bk[1][0].v, s1);
    s1 = wmma_f16(qa[1].v, bk[1][1].v, s1);

    float mnew[8];
#pragma unroll
    for (int r = 0; r < 8; ++r) {
      int qr = qbase + r + rhalf;
      float v0 = s0[r] * scale;
      float v1 = s1[r] * scale;
      if (kb + r16 > qr)      v0 = -1e30f;
      if (kb + 16 + r16 > qr) v1 = -1e30f;
      s0[r] = v0; s1[r] = v1;
      float m = fmaxf(v0, v1);
      for (int o = 8; o >= 1; o >>= 1) m = fmaxf(m, __shfl_xor(m, o, 32));
      mnew[r] = fmaxf(mrow[r], m);
    }
#pragma unroll
    for (int r = 0; r < 8; ++r) {
      float corr = __expf(mrow[r] - mnew[r]);
      mrow[r] = mnew[r];
      float p0 = __expf(s0[r] - mnew[r]);
      float p1 = __expf(s1[r] - mnew[r]);
      s0[r] = p0; s1[r] = p1;
      float ps = p0 + p1;
      for (int o = 8; o >= 1; o >>= 1) ps += __shfl_xor(ps, o, 32);
      lrow[r] = lrow[r] * corr + ps;
#pragma unroll
      for (int nt = 0; nt < 4; ++nt) acc[nt][r] *= corr;
    }

    // P (C layout) -> LDS -> A-fragment layout. Per-wave region, no block
    // barrier (trip counts diverge per wave); LDS is in-order per wave, wait
    // on the split DS counter before re-reading.
    _Float16* P = sP[wave];
#pragma unroll
    for (int r = 0; r < 8; ++r) {
      int pr = r + rhalf;
      P[pr * 32 + r16]      = (_Float16)s0[r];
      P[pr * 32 + 16 + r16] = (_Float16)s1[r];
    }
    asm volatile("s_wait_dscnt 0" ::: "memory");
    Frag pa;
    pa.q[0] = *(const uint4*)&P[r16 * 32 + aklo];
    pa.q[1] = *(const uint4*)&P[r16 * 32 + 16 + aklo];

#pragma unroll
    for (int nt = 0; nt < 4; ++nt) {
      const _Float16* vrow = Vb + (size_t)(nt * 16 + r16) * S_ + kb;
      Frag bv;
      bv.q[0] = *(const uint4*)&vrow[aklo];
      bv.q[1] = *(const uint4*)&vrow[16 + aklo];
      acc[nt] = wmma_f16(pa.v, bv.v, acc[nt]);
    }
  }

#pragma unroll
  for (int nt = 0; nt < 4; ++nt)
#pragma unroll
    for (int r = 0; r < 8; ++r) {
      int qr = qbase + r + rhalf;
      Out[((size_t)b * S_ + qr) * D_ + h * DK_ + nt * 16 + r16] =
          (_Float16)(acc[nt][r] / lrow[r]);
    }
}

// ---------------------------------------------------------------------------
extern "C" void kernel_launch(void* const* d_in, const int* in_sizes, int n_in,
                              void* d_out, int out_size, void* d_ws, size_t ws_size,
                              hipStream_t stream) {
  const float* q    = (const float*)d_in[0];
  const float* k    = (const float*)d_in[1];
  const float* v    = (const float*)d_in[2];
  // d_in[3] = causal mask (int) — applied analytically in flash_attn
  const float* cosT = (const float*)d_in[4];
  const float* sinT = (const float*)d_in[5];
  const float* wq   = (const float*)d_in[6];
  const float* bq   = (const float*)d_in[7];
  const float* wk   = (const float*)d_in[8];
  const float* bk   = (const float*)d_in[9];
  const float* wv   = (const float*)d_in[10];
  const float* bv   = (const float*)d_in[11];
  const float* wo   = (const float*)d_in[12];
  const float* bo   = (const float*)d_in[13];
  const float* qnw  = (const float*)d_in[14];
  const float* knw  = (const float*)d_in[15];

  char* ws = (char*)d_ws;
  const size_t MB = 1ull << 20;
  _Float16* qh16 = (_Float16*)(ws + 0 * MB);    // 8 MB (later reused as Attn)
  _Float16* kh16 = (_Float16*)(ws + 8 * MB);    // 8 MB
  _Float16* vh16 = (_Float16*)(ws + 16 * MB);   // 8 MB
  _Float16* wqh  = (_Float16*)(ws + 24 * MB);   // 2 MB
  _Float16* wkh  = (_Float16*)(ws + 26 * MB);   // 2 MB
  _Float16* wvh  = (_Float16*)(ws + 28 * MB);   // 2 MB
  _Float16* woh  = (_Float16*)(ws + 30 * MB);   // 2 MB
  float*    Qf   = (float*)(ws + 32 * MB);      // 16 MB
  float*    Kf   = (float*)(ws + 48 * MB);      // 16 MB
  float*    Vf   = (float*)(ws + 64 * MB);      // 16 MB
  _Float16* Qh   = (_Float16*)(ws + 80 * MB);   // 8 MB
  _Float16* Kh   = (_Float16*)(ws + 88 * MB);   // 8 MB
  _Float16* Vt   = (_Float16*)(ws + 96 * MB);   // 8 MB
  _Float16* Attn = qh16;  // qh16 fully consumed by GEMM-Q before flash_attn

  const int M = B_ * S_;       // 4096
  const int nIn4 = (M * D_) / 4;      // q/k/v element count / 4
  const int nW4  = (D_ * D_) / 4;     // weight element count / 4

  cvt_f32_to_f16<<<(nIn4 + 255) / 256, 256, 0, stream>>>(q,  qh16, nIn4);
  cvt_f32_to_f16<<<(nIn4 + 255) / 256, 256, 0, stream>>>(k,  kh16, nIn4);
  cvt_f32_to_f16<<<(nIn4 + 255) / 256, 256, 0, stream>>>(v,  vh16, nIn4);
  cvt_f32_to_f16<<<(nW4 + 255) / 256, 256, 0, stream>>>(wq, wqh, nW4);
  cvt_f32_to_f16<<<(nW4 + 255) / 256, 256, 0, stream>>>(wk, wkh, nW4);
  cvt_f32_to_f16<<<(nW4 + 255) / 256, 256, 0, stream>>>(wv, wvh, nW4);
  cvt_f32_to_f16<<<(nW4 + 255) / 256, 256, 0, stream>>>(wo, woh, nW4);

  dim3 grid(M / 128, D_ / 128);
  gemm_f16_wmma<<<grid, 256, 0, stream>>>(qh16, wqh, bq, Qf, M, D_, D_);
  gemm_f16_wmma<<<grid, 256, 0, stream>>>(kh16, wkh, bk, Kf, M, D_, D_);
  gemm_f16_wmma<<<grid, 256, 0, stream>>>(vh16, wvh, bv, Vf, M, D_, D_);
  norm_rope_pack<<<B_ * S_ * H_, 32, 0, stream>>>(Qf, Kf, Vf, cosT, sinT,
                                                  qnw, knw, Qh, Kh, Vt);
  flash_attn<<<(B_ * H_ * (S_ / 16)) / 8, 256, 0, stream>>>(Qh, Kh, Vt, Attn);
  gemm_f16_wmma<<<grid, 256, 0, stream>>>(Attn, woh, bo, (float*)d_out,
                                          M, D_, D_);
}
